// MultiTimeAttention_32375463477366
// MI455X (gfx1250) — matrix operand order repaired and checked by reading, not compiled, not run on hardware
//
#include <hip/hip_runtime.h>
#include <hip/hip_bf16.h>

// ---------------- problem constants (from reference) ----------------
constexpr int Bc  = 16;
constexpr int Hc  = 4;
constexpr int REFc = 128;
constexpr int KEYc = 256;
constexpr int DIMc = 32;
constexpr int ETc  = 128;
constexpr int NHc  = 128;
constexpr int ETKc = ETc / Hc;   // 32

typedef __attribute__((ext_vector_type(16))) _Float16 v16h;
typedef __attribute__((ext_vector_type(8)))  _Float16 v8h;
typedef __attribute__((ext_vector_type(8)))  float    v8f;

// ---------------- WMMA fragment loaders (wave32, 16x16x32 f16) ------
// A (16x32, row-major source): lane = m (0..15) + 16*hh.
//   element e -> K = k0 + 16*hh + e  (contiguous 32B)
__device__ __forceinline__ v16h load_a_frag(const _Float16* rowptr, int k0, int hh) {
    return *(const v16h*)(rowptr + k0 + 16 * hh);
}

// B (32x16) loaded from a TRANSPOSED (N x K row-major) source:
//   lane = n (0..15) + 16*hh, element e -> K = k0 + 16*hh + e  (contiguous 32B)
__device__ __forceinline__ v16h load_b_frag(const _Float16* colptr, int k0, int hh) {
    return *(const v16h*)(colptr + k0 + 16 * hh);
}

__device__ __forceinline__ v8f wmma32(v16h a, v16h b, v8f c) {
    return __builtin_amdgcn_wmma_f32_16x16x32_f16(false, a, false, b, (short)0, c,
                                                  false, false);
}

// ---------------- prep kernels --------------------------------------
__global__ void k_cvt_f16(const float* __restrict__ src, _Float16* __restrict__ dst, int n) {
    int i = blockIdx.x * blockDim.x + threadIdx.x;
    if (i < n) dst[i] = (_Float16)src[i];
}

// dst[c][r] = (f16) src[r][c]   (rows x cols f32 -> cols x rows f16)
__global__ void k_transpose_f16(const float* __restrict__ src, _Float16* __restrict__ dst,
                                int rows, int cols) {
    int i = blockIdx.x * blockDim.x + threadIdx.x;
    if (i < rows * cols) {
        int r = i / cols, c = i % cols;
        dst[c * rows + r] = (_Float16)src[i];
    }
}

// mvT[b][d][k] = mask ? value : 0 ;  mT[b][d][k] = mask ? 1 : 0
__global__ void k_build_mv(const float* __restrict__ value, const int* __restrict__ mask,
                           _Float16* __restrict__ mvT, _Float16* __restrict__ mT) {
    int i = blockIdx.x * blockDim.x + threadIdx.x;
    int total = Bc * KEYc * DIMc;
    if (i >= total) return;
    int b = i / (KEYc * DIMc);
    int rem = i % (KEYc * DIMc);
    int k = rem / DIMc;
    int d = rem % DIMc;
    int m = mask[i];
    size_t o = (size_t)b * DIMc * KEYc + (size_t)d * KEYc + k;
    mvT[o] = m ? (_Float16)value[i] : (_Float16)0.f;
    mT[o]  = (_Float16)(m ? 1.f : 0.f);
}

// ---------------- projection GEMM (1 wave per 16x16 C tile) ---------
// C[b] (M x 128, f16) = A[b](M x 128, f16) @ W(128 x 128, via WT) + bias
__global__ void k_gemm_bias_f16(const _Float16* __restrict__ A, const _Float16* __restrict__ BT,
                                const float* __restrict__ bias, _Float16* __restrict__ C,
                                int lda, int ldb, int ldc, int kTot,
                                long strideA, long strideC) {
    int lane = threadIdx.x;
    int hh = lane >> 4, l16 = lane & 15;
    int m0 = blockIdx.x * 16, n0 = blockIdx.y * 16, bz = blockIdx.z;
    const _Float16* rowp = A + (size_t)bz * strideA + (size_t)(m0 + l16) * lda;
    const _Float16* colp = BT + (size_t)(n0 + l16) * ldb;
    v8f acc = {};
#pragma unroll 4
    for (int k0 = 0; k0 < kTot; k0 += 32)
        acc = wmma32(load_a_frag(rowp, k0, hh), load_b_frag(colp, k0, hh), acc);
    _Float16* Cb = C + (size_t)bz * strideC;
    float bb = bias[n0 + l16];
#pragma unroll
    for (int j = 0; j < 8; ++j)
        Cb[(size_t)(m0 + j + 8 * hh) * ldc + n0 + l16] = (_Float16)(acc[j] + bb);
}

// ---------------- fused attention kernel ----------------------------
// One block (8 waves, 256 threads) handles 16 query rows of one (b,h):
//  phase 1: wave w computes score tile rows m0..m0+15, cols w*32..w*32+31
//           via WMMA, scales by 1/sqrt(32), stages f32 scores in LDS.
//  phase 2: cooperative row-max over all 256 keys + exp -> E (f16) in LDS.
//           (channel-independent max is valid: softmax is shift-invariant)
//  phase 3: waves 0..1 compute N = E@(M(.)V), D = E@M with WMMA
//           (A-fragments from LDS), X = N/D merged-head output (f16).
__global__ void k_attn(const _Float16* __restrict__ qp, const _Float16* __restrict__ kp,
                       const _Float16* __restrict__ mvT, const _Float16* __restrict__ mT,
                       _Float16* __restrict__ X) {
    __shared__ __align__(32) float    S_lds[16][KEYc];      // 16 KB
    __shared__ __align__(32) _Float16 E_lds[16][KEYc];      //  8 KB

    int tid  = threadIdx.x;
    int wv   = tid >> 5;
    int lane = tid & 31;
    int hh   = lane >> 4, l16 = lane & 15;
    int m0 = blockIdx.x * 16;
    int bh = blockIdx.z, b = bh >> 2, h = bh & 3;

    // ---- phase 1: scores ----
    {
        int n0 = wv * 32;
        const _Float16* rowp  = qp + ((size_t)b * REFc + m0 + l16) * ETc + h * ETKc;
        const _Float16* colp0 = kp + ((size_t)b * KEYc + n0 + l16) * ETc + h * ETKc;
        const _Float16* colp1 = kp + ((size_t)b * KEYc + n0 + 16 + l16) * ETc + h * ETKc;
        v16h a = load_a_frag(rowp, 0, hh);
        v8f s0 = {}, s1 = {};
        s0 = wmma32(a, load_b_frag(colp0, 0, hh), s0);
        s1 = wmma32(a, load_b_frag(colp1, 0, hh), s1);
        const float scale = 0.17677669529663687f;   // 1/sqrt(32)
#pragma unroll
        for (int j = 0; j < 8; ++j) {
            int m = j + 8 * hh;
            S_lds[m][n0 + l16]      = s0[j] * scale;
            S_lds[m][n0 + 16 + l16] = s1[j] * scale;
        }
    }
    __syncthreads();

    // ---- phase 2: row max + exp ----
    {
        int row = tid >> 4;       // 16 threads per row (contiguous half-wave)
        int c0  = tid & 15;
        float mx = -1e30f;
#pragma unroll
        for (int c = c0; c < KEYc; c += 16) mx = fmaxf(mx, S_lds[row][c]);
#pragma unroll
        for (int msk = 1; msk <= 8; msk <<= 1) mx = fmaxf(mx, __shfl_xor(mx, msk, 32));
#pragma unroll
        for (int c = c0; c < KEYc; c += 16)
            E_lds[row][c] = (_Float16)__expf(S_lds[row][c] - mx);
    }
    __syncthreads();

    // ---- phase 3: context GEMMs (waves 0,1 only; wave-uniform branch) ----
    if (wv < 2) {
        int n0 = wv * 16;                               // over DIM = 32
        const _Float16* rowp = &E_lds[l16][0];          // A row m = l16, lda = 256
        const _Float16* c1 = mvT + ((size_t)b * DIMc + n0 + l16) * KEYc;
        const _Float16* c2 = mT  + ((size_t)b * DIMc + n0 + l16) * KEYc;
        v8f cn = {}, cd = {};
#pragma unroll
        for (int k0 = 0; k0 < KEYc; k0 += 32) {
            v16h a = load_a_frag(rowp, k0, hh);
            cn = wmma32(a, load_b_frag(c1, k0, hh), cn);
            cd = wmma32(a, load_b_frag(c2, k0, hh), cd);
        }
        _Float16* Xb = X + (size_t)b * REFc * NHc;
#pragma unroll
        for (int j = 0; j < 8; ++j) {
            float d = cd[j];
            float x = cn[j] / (d > 0.f ? d : 1.f);
            Xb[(size_t)(m0 + j + 8 * hh) * NHc + h * DIMc + n0 + l16] = (_Float16)x;
        }
    }
}

// out[b] (128 x 128, f32) = X[b] @ Wo + bo
__global__ void k_outproj(const _Float16* __restrict__ X, const _Float16* __restrict__ woT,
                          const float* __restrict__ bo, float* __restrict__ out) {
    int lane = threadIdx.x;
    int hh = lane >> 4, l16 = lane & 15;
    int m0 = blockIdx.x * 16, n0 = blockIdx.y * 16, b = blockIdx.z;
    const _Float16* rowp = X + ((size_t)b * REFc + m0 + l16) * NHc;
    const _Float16* colp = woT + (size_t)(n0 + l16) * NHc;   // woT: NH x (H*DIM)
    v8f acc = {};
#pragma unroll
    for (int k0 = 0; k0 < Hc * DIMc; k0 += 32)
        acc = wmma32(load_a_frag(rowp, k0, hh), load_b_frag(colp, k0, hh), acc);
    float bb = bo[n0 + l16];
    float* ob = out + (size_t)b * REFc * NHc;
#pragma unroll
    for (int j = 0; j < 8; ++j)
        ob[(size_t)(m0 + j + 8 * hh) * NHc + n0 + l16] = acc[j] + bb;
}

// ---------------- host launch ---------------------------------------
extern "C" void kernel_launch(void* const* d_in, const int* in_sizes, int n_in,
                              void* d_out, int out_size, void* d_ws, size_t ws_size,
                              hipStream_t stream) {
    const float* query = (const float*)d_in[0];
    const float* key   = (const float*)d_in[1];
    const float* value = (const float*)d_in[2];
    const int*   mask  = (const int*)d_in[3];
    const float* Wq = (const float*)d_in[4];
    const float* bq = (const float*)d_in[5];
    const float* Wk = (const float*)d_in[6];
    const float* bk = (const float*)d_in[7];
    const float* Wo = (const float*)d_in[8];
    const float* bo = (const float*)d_in[9];
    float* out = (float*)d_out;

    char* ws = (char*)d_ws;
    size_t off = 0;
    auto take = [&](size_t bytes) -> char* {
        char* p = ws + off;
        off = (off + bytes + 255) & ~(size_t)255;
        return p;
    };
    _Float16* qh  = (_Float16*)take((size_t)Bc * REFc * ETc * 2);   // query f16
    _Float16* kh  = (_Float16*)take((size_t)Bc * KEYc * ETc * 2);   // key f16
    _Float16* qp  = (_Float16*)take((size_t)Bc * REFc * ETc * 2);   // Q projected
    _Float16* kp  = (_Float16*)take((size_t)Bc * KEYc * ETc * 2);   // K projected
    _Float16* wqT = (_Float16*)take((size_t)ETc * ETc * 2);
    _Float16* wkT = (_Float16*)take((size_t)ETc * ETc * 2);
    _Float16* woT = (_Float16*)take((size_t)NHc * Hc * DIMc * 2);
    _Float16* mvT = (_Float16*)take((size_t)Bc * DIMc * KEYc * 2);
    _Float16* mT  = (_Float16*)take((size_t)Bc * DIMc * KEYc * 2);
    _Float16* X   = (_Float16*)take((size_t)Bc * REFc * NHc * 2);

    // --- prep: f16 conversions / transposes / masked value ---
    {
        int n = Bc * REFc * ETc;
        k_cvt_f16<<<(n + 255) / 256, 256, 0, stream>>>(query, qh, n);
    }
    {
        int n = Bc * KEYc * ETc;
        k_cvt_f16<<<(n + 255) / 256, 256, 0, stream>>>(key, kh, n);
    }
    k_transpose_f16<<<(ETc * ETc + 255) / 256, 256, 0, stream>>>(Wq, wqT, ETc, ETc);
    k_transpose_f16<<<(ETc * ETc + 255) / 256, 256, 0, stream>>>(Wk, wkT, ETc, ETc);
    k_transpose_f16<<<(Hc * DIMc * NHc + 255) / 256, 256, 0, stream>>>(Wo, woT, Hc * DIMc, NHc);
    {
        int n = Bc * KEYc * DIMc;
        k_build_mv<<<(n + 255) / 256, 256, 0, stream>>>(value, mask, mvT, mT);
    }

    // --- Q / K projections ---
    k_gemm_bias_f16<<<dim3(REFc / 16, ETc / 16, Bc), 32, 0, stream>>>(
        qh, wqT, bq, qp, ETc, ETc, ETc, ETc,
        (long)REFc * ETc, (long)REFc * ETc);
    k_gemm_bias_f16<<<dim3(KEYc / 16, ETc / 16, Bc), 32, 0, stream>>>(
        kh, wkT, bk, kp, ETc, ETc, ETc, ETc,
        (long)KEYc * ETc, (long)KEYc * ETc);

    // --- fused scores + softmax + context ---
    k_attn<<<dim3(REFc / 16, 1, Bc * Hc), 256, 0, stream>>>(qp, kp, mvT, mT, X);

    // --- output projection ---
    k_outproj<<<dim3(REFc / 16, NHc / 16, Bc), 32, 0, stream>>>(X, woT, bo, out);
}
